// CRF_12979391169081
// MI455X (gfx1250) — matrix operand order repaired and checked by reading, not compiled, not run on hardware
//
#include <hip/hip_runtime.h>
#include <hip/hip_bf16.h>

typedef __attribute__((ext_vector_type(16))) _Float16 v16h;
typedef __attribute__((ext_vector_type(8)))  float    v8f;

// Problem constants (fixed by the reference).
#define TAGS   52          // TAGSET + START + STOP
#define TP     64          // tags padded to 64 for WMMA tiles
#define BATCH  128
#define SEQ    256
#define START_ (TAGS - 2)  // 50
#define STOP_  (TAGS - 1)  // 51
#define ST     (SEQ * TAGS)   // 13312, feats stride per batch
#define NEG_BIG (-1.0e30f)
#define LN2F   0.69314718055994531f

#define CHUNKS 208         // 16 rows * 13 b128 chunks (52 floats) per step

// Async global->LDS copy (GLOBAL_LOAD_ASYNC_TO_LDS_B128, ASYNCcnt-tracked).
__device__ __forceinline__ void async_g2l_b128(unsigned lds_off, const void* g) {
    unsigned long long ga = (unsigned long long)(uintptr_t)g;
    asm volatile("global_load_async_to_lds_b128 %0, %1, off"
                 :: "v"(lds_off), "v"(ga) : "memory");
}
__device__ __forceinline__ void wait_async_le(void) {
    asm volatile("s_wait_asynccnt 2" ::: "memory");
}
__device__ __forceinline__ void wait_async_all(void) {
    asm volatile("s_wait_asynccnt 0" ::: "memory");
}

// ---------------------------------------------------------------------------
// Kernel 1: E16[i][j] = exp(trans[i][j]) padded to 64x64 (f16). exp(-10000)->0
// exactly enforces the START/STOP structural zeros.
// ---------------------------------------------------------------------------
__global__ __launch_bounds__(256) void crf_prep(const float* __restrict__ trans,
                                                _Float16* __restrict__ E16) {
    int tid = threadIdx.x;
    for (int k = 0; k < (TP * TP) / 256; ++k) {
        int e = tid + k * 256;
        int i = e / TP, j = e % TP;
        float v = 0.0f;
        if (i < TAGS && j < TAGS) v = __expf(trans[i * TAGS + j]);
        E16[e] = (_Float16)v;
    }
}

// ---------------------------------------------------------------------------
// Kernel 2: forward recursion. One block = 16 batch rows, 4 waves.
// Wave w computes output columns [16w,16w+16); K = 64 (2 chained WMMAs).
// Emissions for step t+1 stream into a double-buffered LDS tile via async
// global->LDS b128 copies, overlapped with exp/WMMA of step t.
// ---------------------------------------------------------------------------
__global__ __launch_bounds__(128) void crf_forward(const float* __restrict__ feats,
                                                   const float* __restrict__ trans,
                                                   const int*   __restrict__ mask,
                                                   const _Float16* __restrict__ E16,
                                                   float* __restrict__ fwd_partial) {
    __shared__ float    part[16][TP];      // log-space partition, pad = NEG_BIG
    __shared__ _Float16 U[16 * TP];        // exp(part - rowmax), A staging
    __shared__ float    emit_s[2][16][TP]; // double-buffered emissions
    __shared__ float    pmax_s[128];
    __shared__ float    rowmax_s[16];
    __shared__ int      mask_s[16];
    __shared__ float    red_s[16];

    const int tid   = threadIdx.x;
    const int wave  = tid >> 5;
    const int lane  = tid & 31;
    const int grp   = lane >> 4;
    const int lan16 = lane & 15;
    const int bm    = blockIdx.x * 16;
    const int n0    = wave * 16;
    const int col   = n0 + lan16;
    const int colc  = col < TAGS ? col : (TAGS - 1);
    const bool colv = (col < TAGS);

    // async-copy chunk assignment: every thread issues exactly 2 chunks
    const int c0 = tid;
    const int c1 = (tid + 128 < CHUNKS) ? (tid + 128) : (CHUNKS - 1);
    const unsigned emit_base = (unsigned)(uintptr_t)&emit_s[0][0][0];

    // ---- init part = feats[b,0,:] + trans[START,:] ----
    #pragma unroll
    for (int k = 0; k < 8; ++k) {
        int e = tid * 8 + k;
        int r = e >> 6, j = e & 63;
        float v = NEG_BIG;
        if (j < TAGS)
            v = feats[(bm + r) * ST + j] + trans[START_ * TAGS + j];
        part[r][j] = v;
    }

    // ---- loop-invariant B fragments (E columns [n0,n0+16)) ----
    v16h b0, b1;
    #pragma unroll
    for (int e = 0; e < 16; ++e) {
        int k0 = grp * 16 + e;
        b0[e] = E16[(k0)      * TP + col];
        b1[e] = E16[(k0 + 32) * TP + col];
    }

    // ---- pre-issue async emissions for t=1 into buffer 1 ----
    {
        int r0 = c0 / 13, q0 = c0 % 13;
        int r1 = c1 / 13, q1 = c1 % 13;
        unsigned base1 = emit_base + (unsigned)(16 * TP * 4);  // buffer 1
        async_g2l_b128(base1 + r0 * TP * 4 + q0 * 16,
                       &feats[(bm + r0) * ST + 1 * TAGS + q0 * 4]);
        async_g2l_b128(base1 + r1 * TP * 4 + q1 * 16,
                       &feats[(bm + r1) * ST + 1 * TAGS + q1 * 4]);
    }
    __syncthreads();

    // ---- each lane's owned part column slice (rows grp*8 .. grp*8+7) ----
    float pcol[8];
    #pragma unroll
    for (int r = 0; r < 8; ++r) pcol[r] = part[grp * 8 + r][col];

    // ---- time recursion ----
    for (int t = 1; t < SEQ; ++t) {
        // (a) partial row maxima
        {
            int rr = tid >> 3;
            int cc = (tid & 7) * 8;
            float m = part[rr][cc];
            #pragma unroll
            for (int j = 1; j < 8; ++j) m = fmaxf(m, part[rr][cc + j]);
            pmax_s[tid] = m;
        }
        __syncthreads();

        // (b) fold row maxima + load step mask (threads 0..15);
        //     all threads: issue async emissions for step t+1 (clamped)
        if (tid < 16) {
            float m = pmax_s[tid * 8];
            #pragma unroll
            for (int s = 1; s < 8; ++s) m = fmaxf(m, pmax_s[tid * 8 + s]);
            rowmax_s[tid] = m;
            mask_s[tid]   = mask[(bm + tid) * SEQ + t];
        }
        {
            int tn = (t + 1 < SEQ) ? (t + 1) : t;
            unsigned nb = (unsigned)((t + 1) & 1) * (16 * TP * 4);
            int r0 = c0 / 13, q0 = c0 % 13;
            int r1 = c1 / 13, q1 = c1 % 13;
            async_g2l_b128(emit_base + nb + r0 * TP * 4 + q0 * 16,
                           &feats[(bm + r0) * ST + tn * TAGS + q0 * 4]);
            async_g2l_b128(emit_base + nb + r1 * TP * 4 + q1 * 16,
                           &feats[(bm + r1) * ST + tn * TAGS + q1 * 4]);
        }
        __syncthreads();

        // (c) U = exp(part - rowmax) in f16 (pad cols underflow to 0)
        #pragma unroll 4
        for (int k = 0; k < 8; ++k) {
            int e = tid * 8 + k;
            int r = e >> 6, j = e & 63;
            U[e] = (_Float16)__expf(part[r][j] - rowmax_s[r]);
        }
        // hoist this lane's row-block maxima/masks (contiguous -> b128 loads)
        float rm8[8];
        int   mk8[8];
        #pragma unroll
        for (int r = 0; r < 8; ++r) rm8[r] = rowmax_s[grp * 8 + r];
        #pragma unroll
        for (int r = 0; r < 8; ++r) mk8[r] = mask_s[grp * 8 + r];
        wait_async_le();     // step-t emissions (2 oldest per wave) landed
        __syncthreads();     // publish every wave's async copies + U

        // (d) A fragments from U (M = lane&15, K contiguous: k = grp*16 + e)
        v16h a0, a1;
        #pragma unroll
        for (int e = 0; e < 16; ++e) {
            int kl = e + grp * 16;
            a0[e] = U[lan16 * TP + kl];
            a1[e] = U[lan16 * TP + 32 + kl];
        }

        // (e) 16x16 tile, K=64: two chained f16->f32 WMMAs
        v8f acc = {};
        acc = __builtin_amdgcn_wmma_f32_16x16x32_f16(
                false, a0, false, b0, (short)0, acc, false, false);
        acc = __builtin_amdgcn_wmma_f32_16x16x32_f16(
                false, a1, false, b1, (short)0, acc, false, false);

        // (f) branchless in-register update; C layout: r -> M=r+grp*8, N=lane&15
        const float* em = &emit_s[t & 1][0][0];
        #pragma unroll
        for (int r = 0; r < 8; ++r) {
            int   M    = r + grp * 8;
            float w    = fmaxf(acc[r], 1.0e-37f);
            float np   = __log2f(w) * LN2F + rm8[r] + em[M * TP + colc];
            float npe  = colv ? np : NEG_BIG;
            float newv = (mk8[r] > 0) ? npe : pcol[r];
            pcol[r]    = newv;
            part[M][col] = newv;
        }
        __syncthreads();
    }
    wait_async_all();

    // ---- final: logsumexp_i(part[r][i] + trans[i][STOP]) per row (f32) ----
    if (tid < 16) {
        float m = NEG_BIG;
        for (int i = 0; i < TAGS; ++i)
            m = fmaxf(m, part[tid][i] + trans[i * TAGS + STOP_]);
        float s = 0.0f;
        for (int i = 0; i < TAGS; ++i)
            s += __expf(part[tid][i] + trans[i * TAGS + STOP_] - m);
        red_s[tid] = __logf(s) + m;
    }
    __syncthreads();
    if (tid == 0) {
        float s = 0.0f;
        for (int r = 0; r < 16; ++r) s += red_s[r];   // fixed order
        fwd_partial[blockIdx.x] = s;
    }
}

// ---------------------------------------------------------------------------
// Kernel 3: gold path score, one lane per batch row.
// ---------------------------------------------------------------------------
__global__ __launch_bounds__(128) void crf_gold(const float* __restrict__ feats,
                                                const float* __restrict__ trans,
                                                const int*   __restrict__ mask,
                                                const int*   __restrict__ tags,
                                                float* __restrict__ gold) {
    int b = threadIdx.x;
    float acc = 0.0f;
    int prev = START_;
    int last = START_;
    for (int t = 0; t < SEQ; ++t) {
        int m  = mask[b * SEQ + t];
        int tg = tags[b * SEQ + t];
        float contrib = feats[b * ST + t * TAGS + tg] + trans[prev * TAGS + tg];
        acc += (m > 0) ? contrib : 0.0f;
        prev = (m > 0) ? tg : prev;
        last = (m > 0) ? tg : last;
    }
    acc += trans[last * TAGS + STOP_];
    gold[b] = acc;
}

// ---------------------------------------------------------------------------
// Kernel 4: deterministic final reduction -> scalar NLL.
// ---------------------------------------------------------------------------
__global__ void crf_reduce(const float* __restrict__ fwd_partial,
                           const float* __restrict__ gold,
                           float* __restrict__ out) {
    float f = 0.0f;
    for (int i = 0; i < BATCH / 16; ++i) f += fwd_partial[i];
    float g = 0.0f;
    for (int b = 0; b < BATCH; ++b) g += gold[b];
    out[0] = f - g;
}

// ---------------------------------------------------------------------------
extern "C" void kernel_launch(void* const* d_in, const int* in_sizes, int n_in,
                              void* d_out, int out_size, void* d_ws, size_t ws_size,
                              hipStream_t stream) {
    const float* feats = (const float*)d_in[0];
    const float* trans = (const float*)d_in[1];
    const int*   mask  = (const int*)d_in[2];
    const int*   tags  = (const int*)d_in[3];

    char* ws = (char*)d_ws;
    _Float16* E16         = (_Float16*)ws;                   // 64*64*2 = 8192 B
    float*    fwd_partial = (float*)(ws + 8192);             // 8 floats
    float*    gold        = (float*)(ws + 8192 + 64);        // 128 floats

    crf_prep   <<<1, 256, 0, stream>>>(trans, E16);
    crf_forward<<<BATCH / 16, 128, 0, stream>>>(feats, trans, mask, E16, fwd_partial);
    crf_gold   <<<1, BATCH, 0, stream>>>(feats, trans, mask, tags, gold);
    crf_reduce <<<1, 1, 0, stream>>>(fwd_partial, gold, (float*)d_out);
}
